// BoundaryPredictor4_61280593380117
// MI455X (gfx1250) — matrix operand order repaired and checked by reading, not compiled, not run on hardware
//
#include <hip/hip_runtime.h>
#include <hip/hip_bf16.h>

// ---------------------------------------------------------------------------
// BoundaryPredictor for MI455X (gfx1250, wave32, WMMA).
//
// Roofline: conv GEMM = 8*341*2046*3072*2 ~= 34.4 GFLOP (f16 WMMA path),
// memory = ~196 MB slab reads + 64 MB pooling read + 16 MB pooled write
// ~= 280 MB -> ~12 us @ 23.3 TB/s. GEMM and pooling are the only heavy parts;
// scan/compaction is O(B*L) scalar work.
// ---------------------------------------------------------------------------

#define KERN   3
#define B_     8
#define L_     2048
#define D_     1024
#define CH     341
#define CHP    352                 // channels padded to 22 tiles of 16
#define NTILES 22
#define KD     (KERN * D_)         // 3072 reduction length
#define NSEG   512
#define NEGV   (-10.0f)
#define LCONV  (L_ - KERN + 1)     // 2046 conv outputs per batch
#define MROWS  (B_ * LCONV)        // 16368 (== 16 * 1023 exactly)
#define MTILES (MROWS / 16)        // 1023
#define KHALF  1536                // K staged in two halves
#define ROWH   (KHALF + 8)         // LDS row stride (halfs), bank-spread pad
#define KSTEPS (KHALF / 32)        // 48 WMMA K-steps per half

typedef __attribute__((ext_vector_type(16))) _Float16 v16h;
typedef __attribute__((ext_vector_type(8)))  _Float16 v8h;
typedef __attribute__((ext_vector_type(4)))  _Float16 v4h;
typedef __attribute__((ext_vector_type(8)))  float    v8f;

// workspace layout (bytes)
#define BT_OFF   0
#define BT_BYTES (CHP * KD * 2)            // 2,162,688
#define LG_OFF   (BT_OFF + BT_BYTES)
#define LG_BYTES (B_ * L_ * 4)
#define ST_OFF   (LG_OFF + LG_BYTES)
#define ST_BYTES (B_ * NSEG * 4)
#define CT_OFF   (ST_OFF + ST_BYTES)
#define CT_BYTES (B_ * NSEG * 4)
#define NK_OFF   (CT_OFF + CT_BYTES)
#define LEN_OFF  (NK_OFF + B_ * 4)

// output layout (floats): pooled | num_boundaries | total_positions | short_mask
#define OUT_NB   (B_ * NSEG * D_)          // 4,194,304
#define OUT_TP   (OUT_NB + 1)
#define OUT_SM   (OUT_NB + 2)

// ---------------------------------------------------------------------------
// Kernel 0: repack w1 (Ch,D,K) f32 -> Bt[c][k*D+d] f16, padded channels = 0.
// ---------------------------------------------------------------------------
__global__ void prep_w1(const float* __restrict__ w1, _Float16* __restrict__ Bt) {
  int idx = blockIdx.x * 256 + threadIdx.x;
  if (idx >= CHP * KD) return;
  int c = idx / KD, kd = idx % KD;
  int k = kd >> 10, d = kd & (D_ - 1);
  float v = (c < CH) ? w1[(c * D_ + d) * KERN + k] : 0.0f;
  Bt[idx] = (_Float16)v;
}

// ---------------------------------------------------------------------------
// Kernel 1: fused conv + relu + channel-dot -> logits, via v_wmma_f32_16x16x32_f16.
// One block = one 16-row M-tile; A-slab in LDS (two K halves); 8 waves split
// the 22 N-tiles; per-row logits combined with ds_add_f32.
// ---------------------------------------------------------------------------
__global__ __launch_bounds__(256) void conv_logits(
    const float* __restrict__ hidden, const _Float16* __restrict__ Bt,
    const float* __restrict__ b1, const float* __restrict__ w2,
    const float* __restrict__ b2, float* __restrict__ logits) {
  __shared__ _Float16 Alds[16 * ROWH];   // 49,408 B
  __shared__ float slog[16];

  const int tid   = threadIdx.x;
  const int lane  = tid & 31;
  const int wave  = tid >> 5;
  const int mtile = blockIdx.x;
  const int lrow  = lane & 15;           // M (A) / N (B) lane index
  const int lhi   = lane >> 4;           // lane half selects K group

  if (tid < 16) slog[tid] = 0.0f;

  v8f c8[3] = {};                        // up to 3 N-tiles per wave, persistent
  float lacc[8] = {0.f, 0.f, 0.f, 0.f, 0.f, 0.f, 0.f, 0.f};

  for (int kh = 0; kh < 2; ++kh) {
    // stage half A-slab: rows = conv windows, cols = (k*D + d) slice
    for (int idx = tid * 4; idx < 16 * KHALF; idx += 256 * 4) {
      int r  = idx / KHALF;
      int kk = idx % KHALF;
      int kd = kh * KHALF + kk;
      int k = kd >> 10, d = kd & (D_ - 1);
      int row = mtile * 16 + r;
      int b = row / LCONV, i = row % LCONV;
      const float4 v = *(const float4*)(hidden + (size_t)(b * L_ + i + k) * D_ + d);
      v4h h4;
      h4[0] = (_Float16)v.x; h4[1] = (_Float16)v.y;
      h4[2] = (_Float16)v.z; h4[3] = (_Float16)v.w;
      *(v4h*)(&Alds[r * ROWH + kk]) = h4;
    }
    __syncthreads();

    const _Float16* ap = &Alds[lrow * ROWH + lhi * 8];
#pragma unroll
    for (int j = 0; j < 3; ++j) {
      int nt = wave + 8 * j;             // wave-uniform guard: EXEC stays full
      if (nt < NTILES) {
        const _Float16* bp =
            Bt + (size_t)(nt * 16 + lrow) * KD + (size_t)kh * KHALF + lhi * 8;
        v8f acc = c8[j];
#pragma unroll 4
        for (int k = 0; k < KSTEPS; ++k) {
          // A frag: K groups [base, base+7] and [16+base, 16+base+7], base = lhi*8
          v8h alo = *(const v8h*)(ap + k * 32);
          v8h ahi = *(const v8h*)(ap + k * 32 + 16);
          // B frag: same K groups as A: [base, base+7] and [16+base, 16+base+7]
          v8h blo = *(const v8h*)(bp + k * 32);
          v8h bhi = *(const v8h*)(bp + k * 32 + 16);
          v16h a = __builtin_shufflevector(alo, ahi, 0, 1, 2, 3, 4, 5, 6, 7,
                                           8, 9, 10, 11, 12, 13, 14, 15);
          v16h bb = __builtin_shufflevector(blo, bhi, 0, 1, 2, 3, 4, 5, 6, 7,
                                            8, 9, 10, 11, 12, 13, 14, 15);
          acc = __builtin_amdgcn_wmma_f32_16x16x32_f16(
              false, a, false, bb, (short)0, acc, false, false);
        }
        c8[j] = acc;
      }
    }
    __syncthreads();
  }

  // relu(h + b1) * w2, accumulated per lane across this wave's N-tiles
#pragma unroll
  for (int j = 0; j < 3; ++j) {
    int nt = wave + 8 * j;
    if (nt < NTILES) {
      int c = nt * 16 + lrow;
      float b1v = (c < CH) ? b1[c] : 0.0f;
      float w2v = (c < CH) ? w2[c] : 0.0f;
#pragma unroll
      for (int r = 0; r < 8; ++r) {
        float h = c8[j][r] + b1v;
        h = h > 0.0f ? h : 0.0f;
        lacc[r] += h * w2v;
      }
    }
  }
  // reduce the 16 channel-lanes in each lane-half; C layout: M = r + 8*lhi
#pragma unroll
  for (int r = 0; r < 8; ++r) {
    float v = lacc[r];
    v += __shfl_xor(v, 1);
    v += __shfl_xor(v, 2);
    v += __shfl_xor(v, 4);
    v += __shfl_xor(v, 8);
    if (lrow == 0) atomicAdd(&slog[r + 8 * lhi], v);   // ds_add_f32
  }
  __syncthreads();

  if (tid < 16) {
    int row = mtile * 16 + tid;
    int b = row / LCONV, i = row % LCONV;
    logits[b * L_ + i + (KERN - 1)] = slog[tid] + b2[0];
  }
}

// ---------------------------------------------------------------------------
// Kernel 2: hard boundaries, forced last-real boundary, segment scan,
// short_mask (fully rewritten each call -> deterministic under replay).
// ---------------------------------------------------------------------------
__global__ __launch_bounds__(256) void boundaries(
    const float* __restrict__ logits, const float* __restrict__ mask,
    int* __restrict__ segStart, int* __restrict__ segCnt,
    int* __restrict__ nkeepArr, int* __restrict__ lenArr,
    float* __restrict__ outShort) {
  __shared__ float hard[L_];
  __shared__ int sStart[NSEG];
  __shared__ int sCnt[NSEG];
  __shared__ int sLen, sNk;
  const int b = blockIdx.x, tid = threadIdx.x;

  if (tid == 0) sLen = 0;
  for (int s = tid; s < NSEG; s += 256) { sStart[s] = 0; sCnt[s] = 0; }
  __syncthreads();

  int cnt = 0;
  for (int l = tid; l < L_; l += 256) {
    float m  = mask[b * L_ + l];
    float lg = (l >= KERN - 1) ? logits[b * L_ + l] : NEGV;
    hard[l] = (lg > 0.0f && m > 0.0f) ? 1.0f : 0.0f;   // sigmoid(x)>0.5 <=> x>0
    cnt += (m > 0.0f) ? 1 : 0;
  }
  atomicAdd(&sLen, cnt);
  __syncthreads();

  if (tid == 0) {
    int len = sLen;
    if (len > 0 && len < L_) hard[len - 1] = 1.0f;     // boundary at last real token
    int run = 0;                                        // seg = cumsum(hard) - hard
    for (int l = 0; l < L_; ++l) {
      int s = run;
      if (l < len && s < NSEG) {
        if (sCnt[s] == 0) sStart[s] = l;
        sCnt[s] += 1;
      }
      run += (hard[l] > 0.5f) ? 1 : 0;
    }
    sNk = run;
    nkeepArr[b] = run;
    lenArr[b] = len;
  }
  __syncthreads();

  const int nk = sNk;
  for (int s = tid; s < NSEG; s += 256) {
    segStart[b * NSEG + s] = sStart[s];
    segCnt[b * NSEG + s]  = sCnt[s];
    outShort[b * NSEG + s] = (s < nk) ? 1.0f : 0.0f;   // kept positions have mask==1
  }
}

// ---------------------------------------------------------------------------
// Kernel 3: overwrite the two scalar outputs (no cross-replay accumulation).
// ---------------------------------------------------------------------------
__global__ void finalize_scalars(const int* __restrict__ nkeepArr,
                                 const int* __restrict__ lenArr,
                                 float* __restrict__ outNB,
                                 float* __restrict__ outTP) {
  if (threadIdx.x == 0 && blockIdx.x == 0) {
    int nb = 0, tp = 0;
    for (int b = 0; b < B_; ++b) { nb += nkeepArr[b]; tp += lenArr[b]; }
    outNB[0] = (float)nb;
    outTP[0] = (float)tp;
  }
}

// ---------------------------------------------------------------------------
// Kernel 4: segment mean-pool (contiguous ranges) + sinusoidal PE.
// ---------------------------------------------------------------------------
__global__ __launch_bounds__(256) void pool_pe(
    const float* __restrict__ hidden, const int* __restrict__ segStart,
    const int* __restrict__ segCnt, float* __restrict__ pooled) {
  const int s = blockIdx.x, b = blockIdx.y, tid = threadIdx.x;
  const int st = segStart[b * NSEG + s];
  const int cn = segCnt[b * NSEG + s];
  const int d0 = tid * 4;

  float4 acc = {0.f, 0.f, 0.f, 0.f};
  const float* base = hidden + (size_t)(b * L_) * D_ + d0;
  for (int t = 0; t < cn; ++t) {
    const float* p = base + (size_t)(st + t) * D_;
    if (t + 1 < cn) __builtin_prefetch(p + D_, 0, 0);   // global_prefetch_b8
    float4 v = *(const float4*)p;
    acc.x += v.x; acc.y += v.y; acc.z += v.z; acc.w += v.w;
  }
  const float scale = (cn > 0) ? 1.0f / ((float)cn + 1e-9f) : 0.0f;

  const float LN10K = 9.21034037197618f;                 // ln(10000)
  const float ang0 = (float)s * __expf(-LN10K * (float)d0 / (float)D_);
  const float ang2 = (float)s * __expf(-LN10K * (float)(d0 + 2) / (float)D_);
  float4 o;
  o.x = acc.x * scale + __sinf(ang0);
  o.y = acc.y * scale + __cosf(ang0);
  o.z = acc.z * scale + __sinf(ang2);
  o.w = acc.w * scale + __cosf(ang2);
  *(float4*)(pooled + (size_t)(b * NSEG + s) * D_ + d0) = o;
}

// ---------------------------------------------------------------------------
extern "C" void kernel_launch(void* const* d_in, const int* in_sizes, int n_in,
                              void* d_out, int out_size, void* d_ws, size_t ws_size,
                              hipStream_t stream) {
  (void)in_sizes; (void)n_in; (void)out_size; (void)ws_size;
  const float* hidden = (const float*)d_in[0];
  const float* mask   = (const float*)d_in[1];
  const float* w1     = (const float*)d_in[2];
  const float* b1     = (const float*)d_in[3];
  const float* w2     = (const float*)d_in[4];
  const float* b2     = (const float*)d_in[5];
  float* out = (float*)d_out;
  char*  ws  = (char*)d_ws;

  _Float16* Bt     = (_Float16*)(ws + BT_OFF);
  float* logits    = (float*)(ws + LG_OFF);
  int* segStart    = (int*)(ws + ST_OFF);
  int* segCnt      = (int*)(ws + CT_OFF);
  int* nkeepArr    = (int*)(ws + NK_OFF);
  int* lenArr      = (int*)(ws + LEN_OFF);

  prep_w1<<<(CHP * KD + 255) / 256, 256, 0, stream>>>(w1, Bt);
  conv_logits<<<MTILES, 256, 0, stream>>>(hidden, Bt, b1, w2, b2, logits);
  boundaries<<<B_, 256, 0, stream>>>(logits, mask, segStart, segCnt,
                                     nkeepArr, lenArr, out + OUT_SM);
  finalize_scalars<<<1, 32, 0, stream>>>(nkeepArr, lenArr, out + OUT_NB, out + OUT_TP);
  pool_pe<<<dim3(NSEG, B_), 256, 0, stream>>>(hidden, segStart, segCnt, out);
}
